// MultiQueryAttention_61847529062435
// MI455X (gfx1250) — matrix operand order repaired and checked by reading, not compiled, not run on hardware
//
#include <hip/hip_runtime.h>
#include <hip/hip_bf16.h>

// ---------------------------------------------------------------------------
// MQA for MI455X (gfx1250): bf16 WMMA everywhere, fused attention with
// shuffle-free 3-pass softmax and non-temporal attn streaming stores.
// ---------------------------------------------------------------------------

typedef __bf16 bf16_t;
typedef __attribute__((ext_vector_type(16))) __bf16 v16bf;
typedef __attribute__((ext_vector_type(8)))  __bf16 v8bf;
typedef __attribute__((ext_vector_type(8)))  float  v8f;

static constexpr int Bc = 2;
static constexpr int Sc = 2048;
static constexpr int Dm = 1024;
static constexpr int Hc = 16;
static constexpr int Dh = 64;

__device__ inline v16bf cat8(v8bf lo, v8bf hi) {
  v16bf r;
#pragma unroll
  for (int i = 0; i < 8; ++i) { r[i] = lo[i]; r[i + 8] = hi[i]; }
  return r;
}

// ---------------------------------------------------------------------------
// Generic tiled GEMM:  C[M,N] = A[M,K] @ W[N,K]^T + bias
// 64x64 tile per 128-thread block (4 waves), K staged in 32-chunks in LDS.
// OUT_MODE: 0 = bf16, [B,H,S,64] permute (Q)   1 = bf16 row-major (K)
//           2 = bf16 transposed [B,64,S] (V)   3 = f32 row-major (final out)
// ---------------------------------------------------------------------------
template <bool A_IS_BF16, int OUT_MODE>
__global__ __launch_bounds__(128) void gemm_wmma_kernel(
    const void* __restrict__ Aptr, const float* __restrict__ W,
    const float* __restrict__ bias, void* __restrict__ Cptr,
    int M, int N, int K) {
  const int tid  = threadIdx.x;
  const int lane = tid & 31;
  const int w    = tid >> 5;
  const int m0   = blockIdx.x * 64;
  const int n0   = blockIdx.y * 64;

  __shared__ __align__(64) bf16_t Asub[64][32];
  __shared__ __align__(64) bf16_t Bsub[64][32];

  v8f acc[4];
#pragma unroll
  for (int nt = 0; nt < 4; ++nt)
#pragma unroll
    for (int i = 0; i < 8; ++i) acc[nt][i] = 0.0f;

  for (int kc = 0; kc < K; kc += 32) {
    __syncthreads();
    // ---- stage A tile (64x32) as bf16 ----
    if (A_IS_BF16) {
      const bf16_t* Ab = (const bf16_t*)Aptr;
#pragma unroll
      for (int i = 0; i < 2; ++i) {
        int cc = tid * 2 + i;           // 256 chunks of 8 bf16
        int r = cc >> 2, c8 = cc & 3;
        v8bf x = *(const v8bf*)(Ab + (size_t)(m0 + r) * K + kc + c8 * 8);
        *(v8bf*)&Asub[r][c8 * 8] = x;
      }
    } else {
      const float* Af = (const float*)Aptr;
#pragma unroll
      for (int i = 0; i < 4; ++i) {
        int cc = tid * 4 + i;           // 512 chunks of float4
        int r = cc >> 3, c4 = cc & 7;
        float4 x = *(const float4*)(Af + (size_t)(m0 + r) * K + kc + c4 * 4);
        bf16_t* d = &Asub[r][c4 * 4];
        d[0] = (bf16_t)x.x; d[1] = (bf16_t)x.y;
        d[2] = (bf16_t)x.z; d[3] = (bf16_t)x.w;
      }
    }
    // ---- stage B tile: Bsub[n][k] = W[n0+n][kc+k] (column-major B) ----
#pragma unroll
    for (int i = 0; i < 4; ++i) {
      int cc = tid * 4 + i;
      int n = cc >> 3, c4 = cc & 7;
      float4 x = *(const float4*)(W + (size_t)(n0 + n) * K + kc + c4 * 4);
      bf16_t* d = &Bsub[n][c4 * 4];
      d[0] = (bf16_t)x.x; d[1] = (bf16_t)x.y;
      d[2] = (bf16_t)x.z; d[3] = (bf16_t)x.w;
    }
    __syncthreads();

    // ---- compute: A fragment per ISA 16-bit 16x32 layout ----
    const int row = w * 16 + (lane & 15);
    const int kb  = (lane >> 4) * 8;    // lanes 0-15: K0..7/16..23; 16-31: +8
    v8bf lo = *(const v8bf*)&Asub[row][kb];
    v8bf hi = *(const v8bf*)&Asub[row][kb + 16];
    v16bf a = cat8(lo, hi);
#pragma unroll
    for (int nt = 0; nt < 4; ++nt) {
      const int nn  = nt * 16 + (lane & 15);
      const int kb2 = (lane >> 4) * 8;  // same K split as A fragment
      v16bf bm = cat8(*(const v8bf*)&Bsub[nn][kb2], *(const v8bf*)&Bsub[nn][kb2 + 16]);
      acc[nt] = __builtin_amdgcn_wmma_f32_16x16x32_bf16(
          false, a, false, bm, (short)0, acc[nt], false, false);
    }
  }

  // ---- epilogue: C layout -> lanes 0-15 rows r, lanes 16-31 rows r+8 ----
#pragma unroll
  for (int nt = 0; nt < 4; ++nt) {
#pragma unroll
    for (int r = 0; r < 8; ++r) {
      const int col  = n0 + nt * 16 + (lane & 15);
      const int mrow = m0 + w * 16 + r + ((lane >> 4) * 8);
      float val = acc[nt][r] + bias[col];
      if (OUT_MODE == 0) {               // Q: [B,H,S,64] bf16
        int b = mrow >> 11, s = mrow & 2047;
        int h = col >> 6,   dh = col & 63;
        ((bf16_t*)Cptr)[(((size_t)(b * Hc + h) * Sc) + s) * Dh + dh] = (bf16_t)val;
      } else if (OUT_MODE == 1) {        // K: row-major bf16 [M,N]
        ((bf16_t*)Cptr)[(size_t)mrow * N + col] = (bf16_t)val;
      } else if (OUT_MODE == 2) {        // V: transposed bf16 [B,64,S]
        int b = mrow >> 11, s = mrow & 2047;
        ((bf16_t*)Cptr)[((size_t)b * Dh + col) * Sc + s] = (bf16_t)val;
      } else {                           // final out: f32 row-major
        ((float*)Cptr)[(size_t)mrow * N + col] = val;
      }
    }
  }
}

// ---------------------------------------------------------------------------
// Fused MQA attention. One block (4 waves) per 16-row Q tile of one (b,h).
// Shuffle-free 3-pass softmax:
//   pass A: per-lane running row max   (pure WMMA + v_max in the loop)
//   pass B: per-lane partial sum-exp   (pure WMMA + exp/add in the loop)
//   pass C: normalize, NT-store attn, stage bf16 P in LDS, ctx += P @ V.
// Scores are recomputed each pass: K (512 KB bf16) is L2-resident, WMMAs are
// essentially free next to the mandatory 512 MB attn write.
// ---------------------------------------------------------------------------
__global__ __launch_bounds__(128) void mqa_attn_kernel(
    const bf16_t* __restrict__ Qb,   // [B,H,S,64]
    const bf16_t* __restrict__ Kb,   // [B,S,64]
    const bf16_t* __restrict__ Vtb,  // [B,64,S]
    float* __restrict__ attn,        // [B,H,S,S]
    bf16_t* __restrict__ ctxb) {     // [B,S,1024]
  const int tid  = threadIdx.x;
  const int lane = tid & 31;
  const int w    = tid >> 5;
  const int bid  = blockIdx.x;
  const int qt   = bid & 127;        // 128 q-tiles per (b,h)
  const int bh   = bid >> 7;
  const int h    = bh & 15;
  const int b    = bh >> 4;
  const int q0   = qt * 16;

  __shared__ float red_m[4][16], red_l[4][16];
  __shared__ float fin_m[16], fin_linv[16];
  __shared__ __align__(64) bf16_t Pbuf[4][16][32];   // per-wave P staging
  __shared__ float CtxRed[4][16][64];

  const int lrow = lane & 15;
  const int kb   = (lane >> 4) * 8;
  const int kb2  = (lane >> 4) * 16;

  // Q A-fragments for dh 0..31 and 32..63 (all waves share the same Q tile)
  const bf16_t* qrow = Qb + (((size_t)(b * Hc + h) * Sc) + q0 + lrow) * Dh;
  v16bf aq0 = *(const v16bf*)(qrow + kb2);
  v16bf aq1 = *(const v16bf*)(qrow + 32 + kb2);

  const bf16_t* Kbase = Kb + (size_t)b * Sc * Dh;
  const float scale = 0.125f;        // 1/sqrt(64)

  // ---------------- pass A: per-lane running max (no shuffles) ------------
  float mx[8];
#pragma unroll
  for (int r = 0; r < 8; ++r) mx[r] = -1e30f;

  for (int t = w; t < 128; t += 4) {
    const bf16_t* krow = Kbase + (size_t)(t * 16 + lrow) * Dh;
    if (t + 4 < 128)
      __builtin_prefetch((const void*)(krow + (size_t)64 * Dh), 0, 1);
    v16bf bk0 = *(const v16bf*)(krow + kb2);
    v16bf bk1 = *(const v16bf*)(krow + 32 + kb2);
    v8f s;
#pragma unroll
    for (int i = 0; i < 8; ++i) s[i] = 0.0f;
    s = __builtin_amdgcn_wmma_f32_16x16x32_bf16(false, aq0, false, bk0, (short)0, s, false, false);
    s = __builtin_amdgcn_wmma_f32_16x16x32_bf16(false, aq1, false, bk1, (short)0, s, false, false);
#pragma unroll
    for (int r = 0; r < 8; ++r) mx[r] = fmaxf(mx[r], s[r]);
  }
  // single cross-lane reduction per reg (within each 16-lane half)
#pragma unroll
  for (int r = 0; r < 8; ++r) {
#pragma unroll
    for (int off = 1; off <= 8; off <<= 1) mx[r] = fmaxf(mx[r], __shfl_xor(mx[r], off));
    mx[r] *= scale;                     // scale is positive: commute with max
  }
  if (lane == 0 || lane == 16) {
    const int rb = (lane >> 4) * 8;
#pragma unroll
    for (int r = 0; r < 8; ++r) red_m[w][rb + r] = mx[r];
  }
  __syncthreads();
  if (tid < 16) {
    float M = red_m[0][tid];
    for (int ww = 1; ww < 4; ++ww) M = fmaxf(M, red_m[ww][tid]);
    fin_m[tid] = M;
  }
  __syncthreads();

  float Mr[8];
#pragma unroll
  for (int r = 0; r < 8; ++r) Mr[r] = fin_m[r + ((lane >> 4) * 8)];

  // ---------------- pass B: per-lane partial sum-exp (no shuffles) --------
  float ls[8];
#pragma unroll
  for (int r = 0; r < 8; ++r) ls[r] = 0.0f;

  for (int t = w; t < 128; t += 4) {
    const bf16_t* krow = Kbase + (size_t)(t * 16 + lrow) * Dh;
    v16bf bk0 = *(const v16bf*)(krow + kb2);
    v16bf bk1 = *(const v16bf*)(krow + 32 + kb2);
    v8f s;
#pragma unroll
    for (int i = 0; i < 8; ++i) s[i] = 0.0f;
    s = __builtin_amdgcn_wmma_f32_16x16x32_bf16(false, aq0, false, bk0, (short)0, s, false, false);
    s = __builtin_amdgcn_wmma_f32_16x16x32_bf16(false, aq1, false, bk1, (short)0, s, false, false);
#pragma unroll
    for (int r = 0; r < 8; ++r) ls[r] += __expf(s[r] * scale - Mr[r]);
  }
#pragma unroll
  for (int r = 0; r < 8; ++r) {
#pragma unroll
    for (int off = 1; off <= 8; off <<= 1) ls[r] += __shfl_xor(ls[r], off);
  }
  if (lane == 0 || lane == 16) {
    const int rb = (lane >> 4) * 8;
#pragma unroll
    for (int r = 0; r < 8; ++r) red_l[w][rb + r] = ls[r];
  }
  __syncthreads();
  if (tid < 16) {
    float L = red_l[0][tid] + red_l[1][tid] + red_l[2][tid] + red_l[3][tid];
    fin_linv[tid] = 1.0f / L;
  }
  __syncthreads();

  float Li[8];
#pragma unroll
  for (int r = 0; r < 8; ++r) Li[r] = fin_linv[r + ((lane >> 4) * 8)];

  // ---------------- pass C: attn NT-store + ctx = P @ V -------------------
  v8f acc[4];
#pragma unroll
  for (int nt = 0; nt < 4; ++nt)
#pragma unroll
    for (int i = 0; i < 8; ++i) acc[nt][i] = 0.0f;

  for (int ch = w; ch < 64; ch += 4) {      // 32-key chunks
#pragma unroll
    for (int sub = 0; sub < 2; ++sub) {
      const int n0 = ch * 32 + sub * 16;
      const bf16_t* krow = Kbase + (size_t)(n0 + lrow) * Dh;
      v16bf bk0 = *(const v16bf*)(krow + kb2);
      v16bf bk1 = *(const v16bf*)(krow + 32 + kb2);
      v8f s;
#pragma unroll
      for (int i = 0; i < 8; ++i) s[i] = 0.0f;
      s = __builtin_amdgcn_wmma_f32_16x16x32_bf16(false, aq0, false, bk0, (short)0, s, false, false);
      s = __builtin_amdgcn_wmma_f32_16x16x32_bf16(false, aq1, false, bk1, (short)0, s, false, false);
#pragma unroll
      for (int r = 0; r < 8; ++r) {
        const int rl = r + ((lane >> 4) * 8);
        float p = __expf(s[r] * scale - Mr[r]) * Li[r];
        // 512 MB streaming output: non-temporal so it doesn't evict K/V in L2
        __builtin_nontemporal_store(
            p, &attn[(((size_t)bh * Sc) + q0 + rl) * Sc + n0 + lrow]);
        Pbuf[w][rl][sub * 16 + lrow] = (bf16_t)p;
      }
    }
    asm volatile("s_wait_dscnt 0" ::: "memory");   // wave-local LDS RAW fence
    v16bf ap = *(const v16bf*)&Pbuf[w][lrow][kb2];
#pragma unroll
    for (int nt = 0; nt < 4; ++nt) {
      const bf16_t* vt = Vtb + ((size_t)b * Dh + nt * 16 + lrow) * Sc +
                         ch * 32 + kb2;
      v16bf bv = *(const v16bf*)vt;
      acc[nt] = __builtin_amdgcn_wmma_f32_16x16x32_bf16(
          false, ap, false, bv, (short)0, acc[nt], false, false);
    }
  }

  // cross-wave partial-ctx reduction
#pragma unroll
  for (int nt = 0; nt < 4; ++nt)
#pragma unroll
    for (int r = 0; r < 8; ++r)
      CtxRed[w][r + ((lane >> 4) * 8)][nt * 16 + lrow] = acc[nt][r];
  __syncthreads();
#pragma unroll
  for (int i = 0; i < 8; ++i) {
    const int e = tid * 8 + i;       // 16*64 = 1024 elements
    const int row = e >> 6, dh = e & 63;
    float sum = CtxRed[0][row][dh] + CtxRed[1][row][dh] +
                CtxRed[2][row][dh] + CtxRed[3][row][dh];
    ctxb[((size_t)(b * Sc) + q0 + row) * Dm + h * Dh + dh] = (bf16_t)sum;
  }
}

// ---------------------------------------------------------------------------
extern "C" void kernel_launch(void* const* d_in, const int* in_sizes, int n_in,
                              void* d_out, int out_size, void* d_ws, size_t ws_size,
                              hipStream_t stream) {
  const float* query = (const float*)d_in[0];
  const float* key_i = (const float*)d_in[1];
  const float* value = (const float*)d_in[2];
  const float* Wq_w  = (const float*)d_in[3];
  const float* Wq_b  = (const float*)d_in[4];
  const float* Wk_w  = (const float*)d_in[5];
  const float* Wk_b  = (const float*)d_in[6];
  const float* Wv_w  = (const float*)d_in[7];
  const float* Wv_b  = (const float*)d_in[8];
  const float* Wo_w  = (const float*)d_in[9];
  const float* Wo_b  = (const float*)d_in[10];

  float* out  = (float*)d_out;
  float* attn = out + (size_t)Bc * Sc * Dm;     // tuple order: (out, attn)

  // workspace layout (bf16), all offsets 256B aligned
  char* ws = (char*)d_ws;
  bf16_t* Qb   = (bf16_t*)(ws);                             //  8 MB [B,H,S,64]
  bf16_t* Kb   = (bf16_t*)(ws + 8388608);                   // 512 KB [B,S,64]
  bf16_t* Vtb  = (bf16_t*)(ws + 8388608 + 524288);          // 512 KB [B,64,S]
  bf16_t* ctxb = (bf16_t*)(ws + 8388608 + 1048576);         //  8 MB [B,S,1024]

  const int M = Bc * Sc;  // 4096
  dim3 blk(128);

  // Q / K / V projections
  gemm_wmma_kernel<false, 0><<<dim3(M / 64, Dm / 64), blk, 0, stream>>>(
      (const void*)query, Wq_w, Wq_b, (void*)Qb, M, Dm, Dm);
  gemm_wmma_kernel<false, 1><<<dim3(M / 64, 1), blk, 0, stream>>>(
      (const void*)key_i, Wk_w, Wk_b, (void*)Kb, M, Dh, Dm);
  gemm_wmma_kernel<false, 2><<<dim3(M / 64, 1), blk, 0, stream>>>(
      (const void*)value, Wv_w, Wv_b, (void*)Vtb, M, Dh, Dm);

  // fused attention: B*H*(S/16) = 4096 blocks
  mqa_attn_kernel<<<dim3(Bc * Hc * (Sc / 16)), blk, 0, stream>>>(
      Qb, Kb, Vtb, attn, ctxb);

  // output projection (f32 result + bias)
  gemm_wmma_kernel<true, 3><<<dim3(M / 64, Dm / 64), blk, 0, stream>>>(
      (const void*)ctxb, Wo_w, Wo_b, (void*)out, M, Dm, Dm);
}
